// LongformerSelfAttentionForBart_46918222741896
// MI455X (gfx1250) — compile-verified, matrix operand order checked by reading, not run on hardware
//
#include <hip/hip_runtime.h>

// Longformer self-attention (BART) for MI455X / gfx1250, wave32 WMMA f16.
// S=4096 B=2 D=768 H=12 DH=64 W=256 -> C=128, N=32, 3C=384 band.

typedef _Float16 h16;
typedef __attribute__((ext_vector_type(16))) _Float16 v16h;
typedef __attribute__((ext_vector_type(8)))  float    v8f;

#define S_  4096
#define Bb  2
#define D_  768
#define H_  12
#define DH_ 64
#define C_  128
#define NC_ 32

__device__ __forceinline__ int kpat(int pr, int hi) {
    // 16-bit A/B fragment K pattern: VGPR0..3 -> K {0,1},{2,3},{4,5},{6,7} (+8 for hi half)
    //                                 VGPR4..7 -> K {16,17}..{22,23}      (+8 for hi half)
    return (pr < 4 ? 2 * pr : 16 + 2 * (pr - 4)) + hi * 8;
}

// Load a 16x32 f16 fragment (A layout; B uses same register shape with "row" = column)
// from row-major memory with leading dimension ld (elements).
__device__ __forceinline__ v16h ld_frag(const h16* p, int ld) {
    int lane = threadIdx.x & 31;
    int row  = lane & 15;
    int hi   = lane >> 4;
    const h16* q = p + (size_t)row * ld;
    v16h f;
#pragma unroll
    for (int pr = 0; pr < 8; ++pr) {
        int k = kpat(pr, hi);
        f[2 * pr]     = q[k];
        f[2 * pr + 1] = q[k + 1];
    }
    return f;
}

__device__ __forceinline__ v8f wmma16(v16h a, v16h b, v8f c) {
    return __builtin_amdgcn_wmma_f32_16x16x32_f16(false, a, false, b, (short)0, c, false, false);
}

// ---------------------------------------------------------------- converts --
__global__ __launch_bounds__(256) void conv_x(const float* __restrict__ query,
                                              h16* __restrict__ xh) {
    int i = blockIdx.x * blockDim.x + threadIdx.x;   // i = (b*S+s)*D + d
    if (i >= Bb * S_ * D_) return;
    int d = i % D_;
    int r = i / D_;
    int s = r % S_;
    int b = r / S_;
    xh[i] = (h16)query[((size_t)s * Bb + b) * D_ + d];
}

__global__ __launch_bounds__(256) void conv_w(const float* __restrict__ Wq,
                                              const float* __restrict__ Wk,
                                              const float* __restrict__ Wv,
                                              const float* __restrict__ Wo,
                                              h16* __restrict__ wt) {
    int i = blockIdx.x * blockDim.x + threadIdx.x;   // i = mat*D*D + n*D + k (transposed)
    if (i >= 4 * D_ * D_) return;
    int mat = i / (D_ * D_);
    int r   = i % (D_ * D_);
    int nn  = r / D_;
    int kk  = r % D_;
    const float* W = (mat == 0) ? Wq : (mat == 1) ? Wk : (mat == 2) ? Wv : Wo;
    float v = W[(size_t)kk * D_ + nn];
    if (mat == 0) v *= 0.125f;   // fold 1/sqrt(DH) into Wq
    wt[i] = (h16)v;
}

// ------------------------------------------------------------- QKV GEMM -----
// M = B*S = 8192 rows of xh, N = 3*768 (q|k|v), K = 768. Wave tile = 16x64.
__global__ __launch_bounds__(256) void qkv_gemm(const h16* __restrict__ xh,
                                                const h16* __restrict__ wt,
                                                const float* __restrict__ bq,
                                                const float* __restrict__ bk,
                                                const float* __restrict__ bv,
                                                h16* __restrict__ qh,
                                                h16* __restrict__ kh,
                                                h16* __restrict__ vh) {
    int wid   = blockIdx.x * (blockDim.x >> 5) + (threadIdx.x >> 5);
    int nslab = wid % 36;                  // 2304 / 64
    int mt    = wid / 36;                  // 512 m-tiles
    int lane  = threadIdx.x & 31;
    int row   = lane & 15, hi = lane >> 4;
    int m0    = mt * 16;
    int nc0   = nslab * 64;
    int mat   = nc0 / D_;                  // 0=q 1=k 2=v (slab never straddles)
    int d0    = nc0 % D_;
    const h16* w  = wt + (size_t)mat * D_ * D_;
    const h16* xb = xh + (size_t)m0 * D_;
    v8f acc[4] = {};
    for (int k0 = 0; k0 < D_; k0 += 32) {
        v16h a = ld_frag(xb + k0, D_);
#pragma unroll
        for (int t = 0; t < 4; ++t) {
            v16h bf = ld_frag(w + (size_t)(d0 + t * 16) * D_ + k0, D_);
            acc[t]  = wmma16(a, bf, acc[t]);
        }
    }
    const float* bias = (mat == 0) ? bq : (mat == 1) ? bk : bv;
    h16*         out  = (mat == 0) ? qh : (mat == 1) ? kh : vh;
    float bscale = (mat == 0) ? 0.125f : 1.0f;
#pragma unroll
    for (int t = 0; t < 4; ++t) {
        int   d  = d0 + t * 16 + row;
        float bb = bias[d] * bscale;
#pragma unroll
        for (int r = 0; r < 8; ++r) {
            int m = m0 + r + hi * 8;
            out[(size_t)m * D_ + d] = (h16)(acc[t][r] + bb);
        }
    }
}

// ---------------------------------------------------------- attention -------
// One block per (b, h, chunk). 8 waves x 16 query rows. Streaming softmax over
// the 3C = 384 key band in 32-key steps. Probs bounce through LDS to convert
// C/D register layout -> A fragment layout.
__global__ __launch_bounds__(256) void attn(const h16* __restrict__ qh,
                                            const h16* __restrict__ kh,
                                            const h16* __restrict__ vh,
                                            const float* __restrict__ kpm,
                                            h16* __restrict__ ctxh) {
    int blk = blockIdx.x;                  // b*H*NC + h*NC + n
    int n   = blk % NC_;
    int t1  = blk / NC_;
    int h   = t1 % H_;
    int b   = t1 / H_;
    int wave = threadIdx.x >> 5;
    int lane = threadIdx.x & 31;
    int row  = lane & 15, hi = lane >> 4;

    __shared__ h16 pls[8][16 * 32];

    int p0 = wave * 16;                    // query base within chunk
    const h16* qbase = qh + ((size_t)(b * S_ + n * C_ + p0)) * D_ + h * DH_;
    v16h a0 = ld_frag(qbase, D_);          // dh 0..31
    v16h a1 = ld_frag(qbase + 32, D_);     // dh 32..63

    v8f   acc[4] = {};
    float mrun[8], lrun[8];
#pragma unroll
    for (int r = 0; r < 8; ++r) { mrun[r] = -1e30f; lrun[r] = 0.f; }

    for (int jt = 0; jt < 12; ++jt) {      // 384 keys / 32
        int jj = jt * 32;
        v8f s[2];
#pragma unroll
        for (int t = 0; t < 2; ++t) {      // two 16-key score tiles
            int  jcol = jj + t * 16 + row; // this lane's key column (B layout)
            int  kpos = n * C_ + jcol - C_;
            bool kv   = (kpos >= 0 && kpos < S_);
            const h16* kb = kh + ((size_t)(b * S_ + (kv ? kpos : 0))) * D_ + h * DH_;
            v16h b0, b1;
#pragma unroll
            for (int pr = 0; pr < 8; ++pr) {
                int k = kpat(pr, hi);
                b0[2 * pr]     = kv ? kb[k]          : (h16)0;
                b0[2 * pr + 1] = kv ? kb[k + 1]      : (h16)0;
                b1[2 * pr]     = kv ? kb[32 + k]     : (h16)0;
                b1[2 * pr + 1] = kv ? kb[32 + k + 1] : (h16)0;
            }
            v8f z = {};
            z = wmma16(a0, b0, z);
            z = wmma16(a1, b1, z);
            s[t] = z;
        }
        // mask: band (j>=p && j<=p+256) & in-bounds & !key_padding
#pragma unroll
        for (int t = 0; t < 2; ++t) {
            int  j    = jj + t * 16 + row;
            int  kpos = n * C_ + j - C_;
            bool inb  = (kpos >= 0 && kpos < S_);
            bool pad  = inb ? (kpm[(size_t)b * S_ + kpos] != 0.f) : false;
#pragma unroll
            for (int r = 0; r < 8; ++r) {
                int  p  = p0 + r + hi * 8;
                bool ok = inb && !pad && (j >= p) && (j <= p + 256);
                if (!ok) s[t][r] = -1e9f;
            }
        }
        // online softmax: row max over this 32-key step
        float tm[8];
#pragma unroll
        for (int r = 0; r < 8; ++r) tm[r] = fmaxf(s[0][r], s[1][r]);
#pragma unroll
        for (int off = 1; off < 16; off <<= 1)
#pragma unroll
            for (int r = 0; r < 8; ++r) tm[r] = fmaxf(tm[r], __shfl_xor(tm[r], off, 32));
        float rs[8];
#pragma unroll
        for (int r = 0; r < 8; ++r) {
            float mn = fmaxf(mrun[r], tm[r]);
            float sc = __expf(mrun[r] - mn);
            acc[0][r] *= sc; acc[1][r] *= sc; acc[2][r] *= sc; acc[3][r] *= sc;
            lrun[r] *= sc;
            s[0][r] = __expf(s[0][r] - mn);
            s[1][r] = __expf(s[1][r] - mn);
            rs[r]   = s[0][r] + s[1][r];
            mrun[r] = mn;
        }
#pragma unroll
        for (int off = 1; off < 16; off <<= 1)
#pragma unroll
            for (int r = 0; r < 8; ++r) rs[r] += __shfl_xor(rs[r], off, 32);
#pragma unroll
        for (int r = 0; r < 8; ++r) lrun[r] += rs[r];

        // probs (C/D layout) -> LDS -> A fragment layout
        h16* pl = pls[wave];
#pragma unroll
        for (int t = 0; t < 2; ++t)
#pragma unroll
            for (int r = 0; r < 8; ++r)
                pl[(r + hi * 8) * 32 + t * 16 + row] = (h16)s[t][r];
        __syncthreads();
        v16h pa;
        const h16* plr = pl + row * 32;
#pragma unroll
        for (int pr = 0; pr < 8; ++pr) {
            int k = kpat(pr, hi);
            pa[2 * pr]     = plr[k];
            pa[2 * pr + 1] = plr[k + 1];
        }
        __syncthreads();

        // ctx += P(16x32) x V(32x64)
#pragma unroll
        for (int t = 0; t < 4; ++t) {
            v16h bv;
#pragma unroll
            for (int pr = 0; pr < 8; ++pr) {
                int kk = kpat(pr, hi);
#pragma unroll
                for (int u = 0; u < 2; ++u) {
                    int  j    = jj + kk + u;
                    int  kpos = n * C_ + j - C_;
                    bool kvv  = (kpos >= 0 && kpos < S_);
                    bv[2 * pr + u] = kvv
                        ? vh[((size_t)(b * S_ + kpos)) * D_ + h * DH_ + t * 16 + row]
                        : (h16)0;
                }
            }
            acc[t] = wmma16(pa, bv, acc[t]);
        }
    }

    // normalize and store context (f16)
#pragma unroll
    for (int r = 0; r < 8; ++r) {
        float inv = 1.0f / lrun[r];
        int   m   = r + hi * 8;
        int   sg  = n * C_ + p0 + m;
        size_t base = ((size_t)(b * S_ + sg)) * D_ + h * DH_;
#pragma unroll
        for (int t = 0; t < 4; ++t)
            ctxh[base + t * 16 + row] = (h16)(acc[t][r] * inv);
    }
}

// ------------------------------------------------------ output projection --
__global__ __launch_bounds__(256) void out_gemm(const h16* __restrict__ ctxh,
                                                const h16* __restrict__ wto,
                                                const float* __restrict__ bo,
                                                float* __restrict__ out) {
    int wid   = blockIdx.x * (blockDim.x >> 5) + (threadIdx.x >> 5);
    int nslab = wid % 12;                  // 768/64
    int mt    = wid / 12;                  // 512
    int lane  = threadIdx.x & 31;
    int row   = lane & 15, hi = lane >> 4;
    int m0 = mt * 16, d0 = nslab * 64;
    const h16* xb = ctxh + (size_t)m0 * D_;
    v8f acc[4] = {};
    for (int k0 = 0; k0 < D_; k0 += 32) {
        v16h a = ld_frag(xb + k0, D_);
#pragma unroll
        for (int t = 0; t < 4; ++t) {
            v16h bf = ld_frag(wto + (size_t)(d0 + t * 16) * D_ + k0, D_);
            acc[t]  = wmma16(a, bf, acc[t]);
        }
    }
#pragma unroll
    for (int t = 0; t < 4; ++t) {
        int   d  = d0 + t * 16 + row;
        float bb = bo[d];
#pragma unroll
        for (int r = 0; r < 8; ++r) {
            int m = m0 + r + hi * 8;       // m = b*S + s
            int b = m / S_, s = m % S_;
            out[((size_t)s * Bb + b) * D_ + d] = acc[t][r] + bb;
        }
    }
}

// ----------------------------------------------------------------- launch --
extern "C" void kernel_launch(void* const* d_in, const int* in_sizes, int n_in,
                              void* d_out, int out_size, void* d_ws, size_t ws_size,
                              hipStream_t stream) {
    (void)in_sizes; (void)n_in; (void)out_size; (void)ws_size;
    const float* query = (const float*)d_in[0];
    const float* kpm   = (const float*)d_in[1];
    const float* Wq = (const float*)d_in[2]; const float* bq = (const float*)d_in[3];
    const float* Wk = (const float*)d_in[4]; const float* bk = (const float*)d_in[5];
    const float* Wv = (const float*)d_in[6]; const float* bv = (const float*)d_in[7];
    const float* Wo = (const float*)d_in[8]; const float* bo = (const float*)d_in[9];

    char*  ws  = (char*)d_ws;
    size_t off = 0;
    auto alloc = [&](size_t bytes) -> void* {
        void* p = ws + off;
        off = (off + bytes + 255) & ~(size_t)255;
        return p;
    };
    const size_t XE = (size_t)Bb * S_ * D_;          // 6,291,456 elems
    h16* xh   = (h16*)alloc(XE * sizeof(h16));
    h16* wt   = (h16*)alloc((size_t)4 * D_ * D_ * sizeof(h16));
    h16* qh   = (h16*)alloc(XE * sizeof(h16));
    h16* kh   = (h16*)alloc(XE * sizeof(h16));
    h16* vh   = (h16*)alloc(XE * sizeof(h16));
    h16* ctxh = (h16*)alloc(XE * sizeof(h16));

    conv_x<<<(int)((XE + 255) / 256), 256, 0, stream>>>(query, xh);
    conv_w<<<(4 * D_ * D_ + 255) / 256, 256, 0, stream>>>(Wq, Wk, Wv, Wo, wt);
    // 512 m-tiles * 36 n-slabs = 18432 waves / 8 per block
    qkv_gemm<<<2304, 256, 0, stream>>>(xh, wt, bq, bk, bv, qh, kh, vh);
    attn<<<Bb * H_ * NC_, 256, 0, stream>>>(qh, kh, vh, kpm, ctxh);
    // 512 m-tiles * 12 n-slabs = 6144 waves / 8 per block
    out_gemm<<<768, 256, 0, stream>>>(ctxh, wt + (size_t)3 * D_ * D_, bo, (float*)d_out);
}
